// TemporalAttention_72370198937807
// MI455X (gfx1250) — matrix operand order repaired and checked by reading, not compiled, not run on hardware
//
#include <hip/hip_runtime.h>
#include <hip/hip_bf16.h>

typedef _Float16 half16 __attribute__((ext_vector_type(16)));
typedef float    float8 __attribute__((ext_vector_type(8)));

// ---------------------------------------------------------------------------
// 2x2 max pool: (1024 images, 64x64) -> (1024, 32x32)
// ---------------------------------------------------------------------------
__global__ void pool2_kernel(const float* __restrict__ in, float* __restrict__ out) {
    int idx = blockIdx.x * blockDim.x + threadIdx.x;      // 1,048,576
    int x = idx & 31, y = (idx >> 5) & 31, img = idx >> 10;
    const float* p = in + (size_t)img * 4096 + (y * 2) * 64 + x * 2;
    out[idx] = fmaxf(fmaxf(p[0], p[1]), fmaxf(p[64], p[65]));
}

// ---------------------------------------------------------------------------
// Depthwise 3x3 (multiplier 2), pad 1, +bias, writes f16 (feeds WMMA pw conv).
// Branch-free: clamped coordinates + 0/1 mask -> unconditional batched loads.
// in: (N, C, H, W) f32 ; w: (2C,1,3,3) ; out: (N, 2C, H, W) f16
// ---------------------------------------------------------------------------
__global__ void dw3x3_kernel(const float* __restrict__ in, const float* __restrict__ w,
                             const float* __restrict__ bias, _Float16* __restrict__ out,
                             int N, int C, int H, int W) {
    size_t idx = (size_t)blockIdx.x * blockDim.x + threadIdx.x;
    int Cout = C * 2;
    size_t total = (size_t)N * Cout * H * W;
    if (idx >= total) return;
    int x  = idx % W;
    int y  = (idx / W) % H;
    int oc = (int)((idx / ((size_t)W * H)) % Cout);
    int ni = (int)(idx / ((size_t)W * H * Cout));
    int ic = oc >> 1;
    const float* ip = in + ((size_t)(ni * C + ic)) * H * W;
    const float* wp = w + oc * 9;
    float acc = bias[oc];
#pragma unroll
    for (int ky = 0; ky < 3; ++ky)
#pragma unroll
        for (int kx = 0; kx < 3; ++kx) {
            int yy = y + ky - 1, xx = x + kx - 1;
            float valid = (yy >= 0 && yy < H && xx >= 0 && xx < W) ? 1.0f : 0.0f;
            int yc = min(max(yy, 0), H - 1);
            int xc = min(max(xx, 0), W - 1);
            acc += valid * wp[ky * 3 + kx] * ip[yc * W + xc];
        }
    out[idx] = (_Float16)acc;
}

// ---------------------------------------------------------------------------
// Weight pre-pack: grouped 1x1 conv weights -> dense block-diagonal 32x64 f16
// matrix laid out directly in WMMA A-fragment order (4 tiles: mt x ks, each
// 32 lanes x 16 halves, CDNA5 ISA 7.12.2 16-bit A 16x32 layout).
// Also packs fused BN epilogue coefficients: coef[oc]=scale, coef[32+oc]=shift.
// grid = 8 x 256 (2048 threads)
// ---------------------------------------------------------------------------
__global__ void pack_w_kernel(const float* __restrict__ w, const float* __restrict__ pwb,
                              const float* __restrict__ gamma, const float* __restrict__ beta,
                              const float* __restrict__ mean, const float* __restrict__ var,
                              _Float16* __restrict__ wpack, float* __restrict__ coef, int G) {
    int tid = blockIdx.x * 256 + threadIdx.x;   // 2048
    int e = tid & 15;
    int l = (tid >> 4) & 31;
    int t = tid >> 9;                           // tile: mt*2+ks
    int mt = t >> 1, ks = t & 1;
    int m    = (mt << 4) + (l & 15);
    int kidx = (ks << 5) + ((e >= 8) ? 16 : 0) + ((l >> 4) << 3) + (((e >> 1) & 3) << 1) + (e & 1);
    int Cin_pg = 64 / G, Cout_pg = 32 / G;
    int gm = m / Cout_pg, gk = kidx / Cin_pg;
    float wv = (gm == gk) ? w[m * Cin_pg + (kidx - gk * Cin_pg)] : 0.0f;
    wpack[tid] = (_Float16)wv;
    if (tid < 32) {
        float s = gamma[tid] * rsqrtf(var[tid] + 1e-5f);
        coef[tid]      = s;
        coef[32 + tid] = pwb[tid] * s + beta[tid] - mean[tid] * s;
    }
}

// ---------------------------------------------------------------------------
// Pointwise conv (dense 32x64 packed GEMM) + BN + ReLU via WMMA.
// One wave computes one 16(M) x 16(N pixels) tile; K = 64 in 2 steps.
// All loads unconditional: A-fragment = one 32B contiguous vector per lane,
// B-fragment = 16 coalesced d16 loads. Epilogue uses packed coef.
// ---------------------------------------------------------------------------
__global__ void pw_bn_relu_kernel(const _Float16* __restrict__ in,
                                  const _Float16* __restrict__ wpack,
                                  const float* __restrict__ coef,
                                  float* __restrict__ out, int P) {
    int wave = (blockIdx.x * blockDim.x + threadIdx.x) >> 5;
    int lane = threadIdx.x & 31;
    int NT = P >> 4;
    int nt = wave % NT;
    int tmp = wave / NT;
    int mt = tmp & 1;
    int img = tmp >> 1;
    int mrow = lane & 15, khalf = lane >> 4;
    int pix = nt * 16 + mrow;

    float8 acc = {};
#pragma unroll
    for (int ks = 0; ks < 2; ++ks) {
        half16 a = *(const half16*)(wpack + ((((mt << 1) | ks) * 32 + lane) << 4));
        half16 b;
        const _Float16* bp = in + ((size_t)(img * 64 + (ks << 5) + (khalf << 3))) * P + pix;
#pragma unroll
        for (int e = 0; e < 16; ++e) b[e] = bp[(size_t)(e + ((e >= 8) ? 8 : 0)) * P];
        acc = __builtin_amdgcn_wmma_f32_16x16x32_f16(false, a, false, b, (short)0, acc,
                                                     false, false);
    }
#pragma unroll
    for (int r = 0; r < 8; ++r) {
        int oc = (mt << 4) + r + (khalf << 3);
        float val = acc[r] * coef[oc] + coef[32 + oc];
        out[((size_t)(img * 32 + oc)) * P + pix] = fmaxf(val, 0.0f);
    }
}

// ---------------------------------------------------------------------------
// Repack q conv output into the torch-view "query" layout (16,8,8,1024) as f16.
// ---------------------------------------------------------------------------
__global__ void qt_pack_kernel(const float* __restrict__ q, _Float16* __restrict__ qt) {
    int f  = blockIdx.x * blockDim.x + threadIdx.x;       // 1,048,576
    int p  = f & 1023;
    int i2 = (f >> 10) & 7;
    int i1 = (f >> 13) & 7;
    int i0 = f >> 16;
    int b = i0 >> 2;
    int t = 2 * (i0 & 3) + (i1 >> 2);
    int k = i1 & 3;
    qt[f] = (_Float16)q[((size_t)((b * 8 + i2) * 32 + t * 4 + k)) * 1024 + p];
}

// ---------------------------------------------------------------------------
// Repack k conv output into the "key_t" layout (16,8,1024,8) as f16 (gather).
// ---------------------------------------------------------------------------
__global__ void kt_pack_kernel(const float* __restrict__ kc, _Float16* __restrict__ kt) {
    int f   = blockIdx.x * blockDim.x + threadIdx.x;      // 1,048,576
    int i3  = f & 7;
    int i2k = (f >> 3) & 1023;
    int i1  = (f >> 13) & 7;
    int i0  = f >> 16;
    int b  = i0 >> 2;
    int t  = 2 * (i0 & 3) + (i1 >> 2);
    int x  = (i1 & 3) * 8 + (i2k >> 7);
    int v  = (i2k >> 4) & 7;
    int kk = (i2k >> 2) & 3;
    int y  = ((i2k & 3) << 3) + i3;
    kt[f] = (_Float16)kc[((size_t)((b * 8 + v) * 32 + t * 4 + kk)) * 1024 + y * 32 + x];
}

// ---------------------------------------------------------------------------
// Q·K^T (8x1024 @ 1024x8) per batch via WMMA, then causal mask (-32767),
// LeakyReLU(0.2), row softmax. One wave per batch (grid=128, block=32).
// LDS tiles are zero-padded once so fragment gathers are unconditional.
// ---------------------------------------------------------------------------
__global__ void attn_qk_kernel(const _Float16* __restrict__ qt, const _Float16* __restrict__ kt,
                               float* __restrict__ Pbuf) {
    __shared__ __align__(16) _Float16 sQ[16 * 32];   // rows 8..15 stay zero
    __shared__ __align__(16) _Float16 sK[32 * 16];   // cols 8..15 stay zero
    int bb   = blockIdx.x;            // 0..127 = (i0*8+i1)
    int lane = threadIdx.x;           // block = 32
    size_t qbase = (size_t)bb * 8 * 1024;
    size_t kbase = (size_t)bb * 1024 * 8;

    for (int i = lane; i < 512; i += 32) { sQ[i] = (_Float16)0; sK[i] = (_Float16)0; }
    __syncthreads();

    int mrow  = lane & 15;
    int khalf = lane >> 4;

    float8 acc = {};
    for (int ks = 0; ks < 32; ++ks) {
        int kb = ks * 32;
        {   // Q tile 8x32 halves, coalesced 16B per lane
            int row = lane >> 2, c8 = (lane & 3) * 8;
            *(float4*)(&sQ[row * 32 + c8]) = *(const float4*)(qt + qbase + row * 1024 + kb + c8);
        }
        {   // K tile 32 rows x 8 valid cols, one 16B row per lane
            *(float4*)(&sK[lane * 16]) = *(const float4*)(kt + kbase + (size_t)(kb + lane) * 8);
        }
        __syncthreads();
        half16 a, b;
#pragma unroll
        for (int e = 0; e < 16; ++e) {
            int kidx = ((e >= 8) ? 16 : 0) + (khalf << 3) + (((e >> 1) & 3) << 1) + (e & 1);
            a[e] = sQ[mrow * 32 + kidx];
        }
#pragma unroll
        for (int e = 0; e < 16; ++e) {
            b[e] = sK[((((e >= 8) ? 16 : 0) + (khalf << 3) + (e & 7)) << 4) + mrow];
        }
        acc = __builtin_amdgcn_wmma_f32_16x16x32_f16(false, a, false, b, (short)0, acc,
                                                     false, false);
        __syncthreads();
    }

    // rows m=0..7 live in lanes 0..15 at VGPR r=m; column n = lane.
    int n = lane & 15;
#pragma unroll
    for (int r = 0; r < 8; ++r) {
        float v = acc[r];
        if (n > r) v = -32767.0f;                  // causal mask fill (torch NEG)
        v = (v >= 0.0f) ? v : 0.2f * v;            // LeakyReLU(0.2)
        float rmax = v;
        for (int off = 8; off; off >>= 1) rmax = fmaxf(rmax, __shfl_xor(rmax, off, 16));
        float e = __expf(v - rmax);                // padded cols n>=8 underflow to 0
        float rsum = e;
        for (int off = 8; off; off >>= 1) rsum += __shfl_xor(rsum, off, 16);
        float p = e / rsum;
        if (lane < 16 && n < 8) Pbuf[bb * 64 + r * 8 + n] = p;
    }
}

// ---------------------------------------------------------------------------
// out = P(8x8) @ value(8x4096): bandwidth-bound weighted row sum, f32 VALU.
// Writes directly into the final double-conv input layout (32,32,64,64).
// grid = 128 batches * 16 pixel tiles, block = 256.
// ---------------------------------------------------------------------------
__global__ void attn_pv_kernel(const float* __restrict__ Pbuf, const float* __restrict__ vconv,
                               float* __restrict__ out2) {
    int bb  = blockIdx.x >> 4;
    int pix = ((blockIdx.x & 15) << 8) + threadIdx.x;
    int i0 = bb >> 3, i1 = bb & 7;
    int bB = i0 >> 2, kk = i0 & 3;
    int tch = 2 * (i0 & 3) + (i1 >> 2);
    int vch = tch * 4 + (i1 & 3);
    float vv[8];
#pragma unroll
    for (int c = 0; c < 8; ++c)
        vv[c] = vconv[((size_t)((bB * 8 + c) * 32 + vch)) * 4096 + pix];
#pragma unroll
    for (int r = 0; r < 8; ++r) {
        float acc = 0.0f;
#pragma unroll
        for (int c = 0; c < 8; ++c) acc += Pbuf[bb * 64 + r * 8 + c] * vv[c];
        out2[((size_t)((bB * 8 + i1) * 32 + r * 4 + kk)) * 4096 + pix] = acc;
    }
}

// ---------------------------------------------------------------------------
// Host-side launch
// ---------------------------------------------------------------------------
extern "C" void kernel_launch(void* const* d_in, const int* in_sizes, int n_in,
                              void* d_out, int out_size, void* d_ws, size_t ws_size,
                              hipStream_t stream) {
    (void)in_sizes; (void)n_in; (void)out_size; (void)ws_size;
    const float* X = (const float*)d_in[0];

    // jax tree-flatten order: X, then params sorted: c, c_k, c_q, c_v;
    // within each: bn1{beta,gamma,mean,var}, bn2{...}, c1{dw_b,dw_w,pw_b,pw_w}, c2{...}
    const int B_C = 1, B_CK = 17, B_CQ = 33, B_CV = 49;
    auto PP = [&](int base, int off) { return (const float*)d_in[base + off]; };

    // workspace partitions (bytes)
    char* ws = (char*)d_ws;
    float*     XP  = (float*)(ws);                         //  4 MiB pooled X
    _Float16*  H16 = (_Float16*)(ws + (4u << 20));         // 16 MiB dw outputs (f16)
    float*     Y   = (float*)(ws + (20u << 20));           // 16 MiB mid conv
    float*     QO  = (float*)(ws + (36u << 20));           //  4 MiB q conv out
    float*     KO  = (float*)(ws + (40u << 20));           //  4 MiB k conv out
    float*     VO  = (float*)(ws + (44u << 20));           // 16 MiB v conv out
    _Float16*  QT  = (_Float16*)(ws + (60u << 20));        //  2 MiB query f16
    _Float16*  KT  = (_Float16*)(ws + (62u << 20));        //  2 MiB key_t f16
    float*     PB  = (float*)(ws + (64u << 20));           // 32 KiB softmax P
    float*     O2  = (float*)(ws + (65u << 20));           // 16 MiB attn out
    _Float16*  WPK = (_Float16*)(ws + (81u << 20));        // 8 x 2048 halves packed A
    float*     CF  = (float*)(ws + (81u << 20) + (64u << 10)); // 8 x 64 floats coef

    // pre-pack all 8 pointwise convs (weights independent of activations)
    struct { int base, woff, boff, g0; int G; } pcv[8] = {
        {B_CQ, 11, 10, 0, 4}, {B_CQ, 15, 14, 4, 4},
        {B_CK, 11, 10, 0, 4}, {B_CK, 15, 14, 4, 4},
        {B_CV, 11, 10, 0, 4}, {B_CV, 15, 14, 4, 4},
        {B_C,  11, 10, 0, 1}, {B_C,  15, 14, 4, 1},
    };
    for (int i = 0; i < 8; ++i) {
        int bnb = pcv[i].g0;   // bn1 at +0, bn2 at +4
        pack_w_kernel<<<8, 256, 0, stream>>>(PP(pcv[i].base, pcv[i].woff),
                                             PP(pcv[i].base, pcv[i].boff),
                                             PP(pcv[i].base, bnb + 1), PP(pcv[i].base, bnb + 0),
                                             PP(pcv[i].base, bnb + 2), PP(pcv[i].base, bnb + 3),
                                             WPK + i * 2048, CF + i * 64, pcv[i].G);
    }

    auto run_double = [&](const float* inp, int H, int W, int pbase, int cid, float* outp) {
        int P = H * W;
        int dwBlocks = (32 * 64 * P + 255) / 256;
        int pwBlocks = (32 * 2 * (P >> 4)) / 8;   // waves / 8 (block = 256)
        dw3x3_kernel<<<dwBlocks, 256, 0, stream>>>(inp, PP(pbase, 9), PP(pbase, 8), H16,
                                                   32, 32, H, W);
        pw_bn_relu_kernel<<<pwBlocks, 256, 0, stream>>>(H16, WPK + cid * 2048,
                                                        CF + cid * 64, Y, P);
        dw3x3_kernel<<<dwBlocks, 256, 0, stream>>>(Y, PP(pbase, 13), PP(pbase, 12), H16,
                                                   32, 32, H, W);
        pw_bn_relu_kernel<<<pwBlocks, 256, 0, stream>>>(H16, WPK + (cid + 1) * 2048,
                                                        CF + (cid + 1) * 64, outp, P);
    };

    // q / k branches share the pooled input
    pool2_kernel<<<4096, 256, 0, stream>>>(X, XP);
    run_double(XP, 32, 32, B_CQ, 0, QO);
    run_double(XP, 32, 32, B_CK, 2, KO);
    // v branch at full resolution
    run_double(X, 64, 64, B_CV, 4, VO);

    // attention
    qt_pack_kernel<<<4096, 256, 0, stream>>>(QO, QT);
    kt_pack_kernel<<<4096, 256, 0, stream>>>(KO, KT);
    attn_qk_kernel<<<128, 32, 0, stream>>>(QT, KT, PB);
    attn_pv_kernel<<<2048, 256, 0, stream>>>(PB, VO, O2);

    // final double conv (heads=1) straight into d_out (layout matches B,V,T,K,H,W)
    run_double(O2, 64, 64, B_C, 6, (float*)d_out);
}
